// CrossViewTransformer_65317862638369
// MI455X (gfx1250) — matrix operand order repaired and checked by reading, not compiled, not run on hardware
//
#include <hip/hip_runtime.h>
#include <hip/hip_bf16.h>

typedef __bf16 bf16_t;
typedef __attribute__((ext_vector_type(16))) __bf16 v16bf;
typedef __attribute__((ext_vector_type(8)))  float  v8f;

static constexpr int Bn  = 4;
static constexpr int Cch = 256;
static constexpr int Hh  = 64;
static constexpr int Ww  = 64;
static constexpr int Nn  = Hh * Ww;   // 4096
static constexpr int Cqk = 32;

#define ATTN_WAVES 4

// ---------------------------------------------------------------------------
// WMMA helpers: CDNA5 v_wmma_f32_16x16x32_bf16, wave32.
// A-matrix 16x32 bf16 layout: lane L -> row M=L%16, half=L/16;
//   elements 0..15 = K = half*16 + 0..15 (contiguous)
// B-matrix 32x16 bf16 layout: lane L -> col = L%16, khalf = L/16;
//   elements 0..15 = K = khalf*16 + 0..15 (contiguous)
// ---------------------------------------------------------------------------
__device__ __forceinline__ v16bf load_A16x32(const bf16_t* base, int lda) {
  const int lane = threadIdx.x & 31;
  const int m  = lane & 15;
  const int hf = lane >> 4;
  const bf16_t* p = base + (size_t)m * lda + hf * 16;
  v16bf a;
  *((uint4*)&a)       = *((const uint4*)p);
  *(((uint4*)&a) + 1) = *((const uint4*)(p + 8));
  return a;
}

__device__ __forceinline__ v16bf load_B32x16(const bf16_t* base, int ldb) {
  const int lane = threadIdx.x & 31;
  const int col = lane & 15;
  const int hf  = lane >> 4;
  const bf16_t* p = base + (size_t)col * ldb + hf * 16;
  v16bf b;
  *((uint4*)&b)       = *((const uint4*)p);
  *(((uint4*)&b) + 1) = *((const uint4*)(p + 8));
  return b;
}

__device__ __forceinline__ v8f wmma_bf16(v16bf a, v16bf b, v8f c) {
  return __builtin_amdgcn_wmma_f32_16x16x32_bf16(false, a, false, b,
                                                 (short)0, c, false, false);
}

// ---------------------------------------------------------------------------
// fp32 -> bf16 elementwise convert (weights)
// ---------------------------------------------------------------------------
__global__ void cvt_bf16_kernel(const float* __restrict__ in,
                                bf16_t* __restrict__ out, int n) {
  int i = blockIdx.x * blockDim.x + threadIdx.x;
  if (i < n) out[i] = (bf16_t)in[i];
}

// ---------------------------------------------------------------------------
// [B][C][N] fp32 -> [B][N][C] bf16 transpose via LDS tile
// grid (N/32, C/32, B), block (32, 8)
// ---------------------------------------------------------------------------
__global__ void transpose_cvt_kernel(const float* __restrict__ in,
                                     bf16_t* __restrict__ out) {
  __shared__ float tile[32][33];
  const int b  = blockIdx.z;
  const int n0 = blockIdx.x * 32;
  const int c0 = blockIdx.y * 32;
  const int tx = threadIdx.x, ty = threadIdx.y;
  const float* ip = in  + (size_t)b * Cch * Nn;
  bf16_t*      op = out + (size_t)b * Nn * Cch;
#pragma unroll
  for (int i = 0; i < 32; i += 8)
    tile[ty + i][tx] = ip[(size_t)(c0 + ty + i) * Nn + n0 + tx];
  __syncthreads();
#pragma unroll
  for (int i = 0; i < 32; i += 8)
    op[(size_t)(n0 + ty + i) * Cch + c0 + tx] = (bf16_t)tile[tx][ty + i];
}

// ---------------------------------------------------------------------------
// Q/K projection: Out[b][n][o] = sum_c X[b][n][c] * W[o][c] + bias[o]
// X: [B][N][256] bf16 (transposed features), W: [32][256] bf16, Out: [B][N][32] bf16
// grid (N/64, Cqk/16, B), block 128 (4 waves, one 16x16 tile each)
// ---------------------------------------------------------------------------
__global__ __launch_bounds__(128, 2)
void proj_qk_kernel(const bf16_t* __restrict__ X, const bf16_t* __restrict__ W,
                    const float* __restrict__ bias, bf16_t* __restrict__ Out) {
  const int wid  = threadIdx.x >> 5;
  const int lane = threadIdx.x & 31;
  const int col  = lane & 15;
  const int hf   = lane >> 4;
  const int b  = blockIdx.z;
  const int o0 = blockIdx.y * 16;
  const int n0 = (blockIdx.x * 4 + wid) * 16;

  const bf16_t* Arow = X + ((size_t)b * Nn + n0) * Cch;
  v8f acc = {};
  for (int kb = 0; kb < Cch; kb += 32) {
    v16bf a = load_A16x32(Arow + kb, Cch);
    v16bf w = load_B32x16(W + (size_t)o0 * Cch + kb, Cch);
    acc = wmma_bf16(a, w, acc);
  }
  const float bo = bias[o0 + col];
  bf16_t* op = Out + ((size_t)b * Nn + n0 + 8 * hf) * Cqk + o0 + col;
#pragma unroll
  for (int r = 0; r < 8; ++r) op[(size_t)r * Cqk] = (bf16_t)(acc[r] + bo);
}

// ---------------------------------------------------------------------------
// V projection, channel-major output: Vout[b][c][m] = sum_k Wv[c][k]*tgt[k][m] + bv[c]
// A = Wv rows (16c x 32k), B columns = Xt rows (pixel m, contiguous k)
// grid (N/64, C/16, B), block 128
// ---------------------------------------------------------------------------
__global__ __launch_bounds__(128, 2)
void proj_v_kernel(const bf16_t* __restrict__ Xt, const bf16_t* __restrict__ Wv,
                   const float* __restrict__ bias, bf16_t* __restrict__ Vout) {
  const int wid  = threadIdx.x >> 5;
  const int lane = threadIdx.x & 31;
  const int col  = lane & 15;
  const int hf   = lane >> 4;
  const int b  = blockIdx.z;
  const int c0 = blockIdx.y * 16;
  const int m0 = (blockIdx.x * 4 + wid) * 16;

  const bf16_t* Arow = Wv + (size_t)c0 * Cch;
  const bf16_t* Bcol = Xt + ((size_t)b * Nn + m0) * Cch;
  v8f acc = {};
  for (int kb = 0; kb < Cch; kb += 32) {
    v16bf a = load_A16x32(Arow + kb, Cch);
    v16bf x = load_B32x16(Bcol + kb, Cch);
    acc = wmma_bf16(a, x, acc);
  }
#pragma unroll
  for (int r = 0; r < 8; ++r) {
    const int c = c0 + 8 * hf + r;
    Vout[((size_t)b * Cch + c) * Nn + m0 + col] = (bf16_t)(acc[r] + bias[c]);
  }
}

// ---------------------------------------------------------------------------
// Flash attention + residual add.
// Each wave owns 16 queries, accumulates 16x256 fp32 (16 v8f tiles).
// Qt/Kt: [B][N][32] bf16, V: [B][256][N] bf16 (all L2-resident, ~10MB total).
// grid (N/(16*ATTN_WAVES), B), block 32*ATTN_WAVES
// ---------------------------------------------------------------------------
__global__ __launch_bounds__(32 * ATTN_WAVES, 1)
void attn_kernel(const bf16_t* __restrict__ Qt, const bf16_t* __restrict__ Kt,
                 const bf16_t* __restrict__ V, const float* __restrict__ src,
                 float* __restrict__ out) {
  __shared__ __align__(16) bf16_t Plds[ATTN_WAVES][16 * 32];
  const int wid  = threadIdx.x >> 5;
  const int lane = threadIdx.x & 31;
  const int col  = lane & 15;
  const int hf   = lane >> 4;
  const int b  = blockIdx.y;
  const int q0 = (blockIdx.x * ATTN_WAVES + wid) * 16;

  const bf16_t* qbase = Qt + ((size_t)b * Nn + q0) * Cqk;
  const bf16_t* kbase = Kt + (size_t)b * Nn * Cqk;
  const bf16_t* vbase = V  + (size_t)b * Cch * Nn;
  bf16_t* pl = &Plds[wid][0];

  const v16bf a_q = load_A16x32(qbase, Cqk);   // full Cqk=32 in one A tile

  v8f acc[16];
#pragma unroll
  for (int t = 0; t < 16; ++t) acc[t] = {};
  float mrow[8], lrow[8];
#pragma unroll
  for (int r = 0; r < 8; ++r) { mrow[r] = -3.0e38f; lrow[r] = 0.f; }

  for (int kb = 0; kb < Nn; kb += 32) {
    // ---- scores: two 16x16 tiles, K-dim = Cqk = 32 in a single WMMA each
    v16bf bk0 = load_B32x16(kbase + (size_t)kb * Cqk, Cqk);
    v16bf bk1 = load_B32x16(kbase + (size_t)(kb + 16) * Cqk, Cqk);
    v8f z0 = {}, z1 = {};
    v8f s0 = wmma_bf16(a_q, bk0, z0);
    v8f s1 = wmma_bf16(a_q, bk1, z1);

    // ---- online softmax. Row M = r + 8*hf lives in VGPR r of lanes of half hf;
    // columns (keys) are spread over the 16 lanes of that half, so xor-reduce
    // with masks 1,2,4,8 gives per-row stats uniform within each half.
    float alph[8];
#pragma unroll
    for (int r = 0; r < 8; ++r) {
      float v = fmaxf(s0[r], s1[r]);
      v = fmaxf(v, __shfl_xor(v, 1));
      v = fmaxf(v, __shfl_xor(v, 2));
      v = fmaxf(v, __shfl_xor(v, 4));
      v = fmaxf(v, __shfl_xor(v, 8));
      const float mnew = fmaxf(mrow[r], v);
      const float a  = __expf(mrow[r] - mnew);
      const float p0 = __expf(s0[r] - mnew);
      const float p1 = __expf(s1[r] - mnew);
      float rs = p0 + p1;
      rs += __shfl_xor(rs, 1);
      rs += __shfl_xor(rs, 2);
      rs += __shfl_xor(rs, 4);
      rs += __shfl_xor(rs, 8);
      mrow[r] = mnew;
      lrow[r] = a * lrow[r] + rs;
      alph[r] = a;
      // stash P (C-layout) into LDS row-major [16 q][32 k] for A-layout reload
      pl[(r + 8 * hf) * 32 + col]      = (bf16_t)p0;
      pl[(r + 8 * hf) * 32 + 16 + col] = (bf16_t)p1;
    }

    // rescale running accumulators by alpha(row)
#pragma unroll
    for (int t = 0; t < 16; ++t)
#pragma unroll
      for (int r = 0; r < 8; ++r) acc[t][r] *= alph[r];

    // wave-local LDS RAW: DS ops are in-order per wave; be explicit anyway
    asm volatile("s_wait_dscnt 0" ::: "memory");
    const v16bf a_p = load_A16x32(pl, 32);

    // ---- acc += P(16x32) x V-block(32 keys x 16 channels) for all 16 ch tiles
#pragma unroll
    for (int t = 0; t < 16; ++t) {
      v16bf bv = load_B32x16(vbase + (size_t)(t * 16) * Nn + kb, Nn);
      acc[t] = wmma_bf16(a_p, bv, acc[t]);
    }
  }

  // ---- epilogue: normalize, add residual src_feat, 128-bit stores
  float linv[8];
#pragma unroll
  for (int r = 0; r < 8; ++r) linv[r] = 1.0f / lrow[r];

  const float* sb = src + (size_t)b * Cch * Nn;
  float*       ob = out + (size_t)b * Cch * Nn;
#pragma unroll
  for (int t = 0; t < 16; ++t) {
    const int c = t * 16 + col;
    const size_t off = (size_t)c * Nn + q0 + 8 * hf;  // 8 consecutive queries
    float4 slo = *(const float4*)(sb + off);
    float4 shi = *(const float4*)(sb + off + 4);
    float4 olo, ohi;
    olo.x = acc[t][0] * linv[0] + slo.x;
    olo.y = acc[t][1] * linv[1] + slo.y;
    olo.z = acc[t][2] * linv[2] + slo.z;
    olo.w = acc[t][3] * linv[3] + slo.w;
    ohi.x = acc[t][4] * linv[4] + shi.x;
    ohi.y = acc[t][5] * linv[5] + shi.y;
    ohi.z = acc[t][6] * linv[6] + shi.z;
    ohi.w = acc[t][7] * linv[7] + shi.w;
    *(float4*)(ob + off)     = olo;
    *(float4*)(ob + off + 4) = ohi;
  }
}

// ---------------------------------------------------------------------------
extern "C" void kernel_launch(void* const* d_in, const int* in_sizes, int n_in,
                              void* d_out, int out_size, void* d_ws, size_t ws_size,
                              hipStream_t stream) {
  const float* src = (const float*)d_in[0];
  const float* tgt = (const float*)d_in[1];
  const float* Wq  = (const float*)d_in[2];
  const float* bq  = (const float*)d_in[3];
  const float* Wk  = (const float*)d_in[4];
  const float* bk  = (const float*)d_in[5];
  const float* Wv  = (const float*)d_in[6];
  const float* bv  = (const float*)d_in[7];
  float* out = (float*)d_out;

  char* p = (char*)d_ws;
  const size_t szX = (size_t)Bn * Nn * Cch * sizeof(bf16_t);  // 8 MB
  const size_t szQ = (size_t)Bn * Nn * Cqk * sizeof(bf16_t);  // 1 MB
  bf16_t* Xs  = (bf16_t*)p;  p += szX;                  // src^T  [B][N][C]
  bf16_t* Xt  = (bf16_t*)p;  p += szX;                  // tgt^T  [B][N][C]
  bf16_t* Vb  = (bf16_t*)p;  p += szX;                  // V      [B][C][N]
  bf16_t* Qt  = (bf16_t*)p;  p += szQ;                  // Q^T    [B][N][32]
  bf16_t* Kt  = (bf16_t*)p;  p += szQ;                  // K^T    [B][N][32]
  bf16_t* Wqb = (bf16_t*)p;  p += (size_t)Cqk * Cch * sizeof(bf16_t);
  bf16_t* Wkb = (bf16_t*)p;  p += (size_t)Cqk * Cch * sizeof(bf16_t);
  bf16_t* Wvb = (bf16_t*)p;  p += (size_t)Cch * Cch * sizeof(bf16_t);

  cvt_bf16_kernel<<<(Cqk * Cch + 255) / 256, 256, 0, stream>>>(Wq, Wqb, Cqk * Cch);
  cvt_bf16_kernel<<<(Cqk * Cch + 255) / 256, 256, 0, stream>>>(Wk, Wkb, Cqk * Cch);
  cvt_bf16_kernel<<<(Cch * Cch + 255) / 256, 256, 0, stream>>>(Wv, Wvb, Cch * Cch);

  transpose_cvt_kernel<<<dim3(Nn / 32, Cch / 32, Bn), dim3(32, 8), 0, stream>>>(src, Xs);
  transpose_cvt_kernel<<<dim3(Nn / 32, Cch / 32, Bn), dim3(32, 8), 0, stream>>>(tgt, Xt);

  proj_qk_kernel<<<dim3(Nn / 64, Cqk / 16, Bn), 128, 0, stream>>>(Xs, Wqb, bq, Qt);
  proj_qk_kernel<<<dim3(Nn / 64, Cqk / 16, Bn), 128, 0, stream>>>(Xt, Wkb, bk, Kt);
  proj_v_kernel <<<dim3(Nn / 64, Cch / 16, Bn), 128, 0, stream>>>(Xt, Wvb, bv, Vb);

  attn_kernel<<<dim3(Nn / (16 * ATTN_WAVES), Bn), 32 * ATTN_WAVES, 0, stream>>>(
      Qt, Kt, Vb, src, out);
}
